// ScaledDotProductAttention_21689584845396
// MI455X (gfx1250) — compile-verified, matrix operand order checked by reading, not run on hardware
//
#include <hip/hip_runtime.h>
#include <hip/hip_bf16.h>

// Shapes from reference: B=2, H=16, S=2048, D=64, T=8.0
#define S_LEN 2048
#define DH    64
#define NHEAD 16
#define NBATCH 2
#define NEGV  (-1e9f)

typedef __attribute__((ext_vector_type(16))) _Float16 v16h;
typedef __attribute__((ext_vector_type(8)))  float    v8f;
typedef __attribute__((ext_vector_type(4)))  float    v4f;

// 16 halves = 32 bytes; q[0] = elements 0..7, q[1] = elements 8..15
union F16Frag { v16h v; uint4 q[2]; };

// One block = 128 threads = 4 waves; covers 64 query rows (16 per wave).
// grid.x = S/64 query blocks, grid.y = B*H heads.
__global__ __launch_bounds__(128)
void fa_fwd_attn_kernel(const float* __restrict__ Q,
                        const float* __restrict__ K,
                        const float* __restrict__ V,
                        const int*   __restrict__ Mask,
                        float* __restrict__ Out,
                        float* __restrict__ Attn)
{
    __shared__ _Float16 Qh[64 * DH];     // 8 KB: query tile, prescaled by 1/T
    __shared__ _Float16 Kh[32 * DH];     // 4 KB: pair of key tiles [key][d]
    __shared__ _Float16 VTh[DH * 32];    // 4 KB: V transposed      [d][key]
    __shared__ _Float16 Ph[4][16 * 32];  // 4 KB: per-wave P re-layout scratch

    const int tid  = threadIdx.x;
    const int wave = tid >> 5;
    const int lane = tid & 31;
    const int l16  = lane & 15;
    const int hi   = lane >> 4;              // 0: lanes 0-15, 1: lanes 16-31

    const int bh = blockIdx.y;
    const int b  = bh >> 4;                  // H == 16
    const int r0 = blockIdx.x * 64;          // first query row of this block
    const int m0 = r0 + wave * 16;           // first query row of this wave

    const float* q  = Q + (size_t)bh * S_LEN * DH;
    const float* kp = K + (size_t)bh * S_LEN * DH;
    const float* vp = V + (size_t)bh * S_LEN * DH;
    const int*   am = Mask + (size_t)b * S_LEN;
    float* outp  = Out  + (size_t)bh * S_LEN * DH;
    float* attnp = Attn + (size_t)bh * S_LEN * S_LEN;

    // ---- stage Q tile (64 x 64) into LDS as f16, prescaled by 1/T=0.125 ----
    for (int idx = tid; idx < 64 * 16; idx += 128) {
        const int row = idx >> 4;
        const int c4  = (idx & 15) << 2;
        const float4 f = *(const float4*)(q + (size_t)(r0 + row) * DH + c4);
        _Float16* d = &Qh[row * DH + c4];
        d[0] = (_Float16)(f.x * 0.125f);
        d[1] = (_Float16)(f.y * 0.125f);
        d[2] = (_Float16)(f.z * 0.125f);
        d[3] = (_Float16)(f.w * 0.125f);
    }
    __syncthreads();

    // ---- per-wave Q A-fragments, ISA 16-bit A 16x32 layout ----
    // lane row = m-local (l16); element i -> d = (i&7) + (i>=8?16:0) + hi*8 + 32*c
    F16Frag qf[2];
    {
        const _Float16* qrow = &Qh[(wave * 16 + l16) * DH];
        #pragma unroll
        for (int c = 0; c < 2; ++c) {
            qf[c].q[0] = *(const uint4*)(qrow + c * 32 + hi * 8);
            qf[c].q[1] = *(const uint4*)(qrow + c * 32 + hi * 8 + 16);
        }
    }

    const int npairs = (r0 + 63) / 32 + 1;   // 32-key pairs up to the diagonal
    const int mlast  = m0 + 15;              // this wave's last query row

    // ================= pass A: streaming row max / row sum =================
    float rmax[8], rsum[8];
    #pragma unroll
    for (int r = 0; r < 8; ++r) { rmax[r] = -3.0e38f; rsum[r] = 0.f; }

    for (int p = 0; p < npairs; ++p) {
        const int kb = p * 32;
        __syncthreads();                      // Kh reuse guard
        for (int idx = tid; idx < 32 * 16; idx += 128) {
            const int row = idx >> 4;
            const int c4  = (idx & 15) << 2;
            const float4 f = *(const float4*)(kp + (size_t)(kb + row) * DH + c4);
            _Float16* d = &Kh[row * DH + c4];
            d[0] = (_Float16)f.x; d[1] = (_Float16)f.y;
            d[2] = (_Float16)f.z; d[3] = (_Float16)f.w;
        }
        if (p + 1 < npairs)  // warm L2/L0 for the next key pair
            __builtin_prefetch(kp + (size_t)(kb + 32) * DH + tid * 16, 0, 3);
        __syncthreads();

        if (kb > mlast) continue;             // fully above the diagonal: no effect

        // ---- both 16-key subtiles, then one merged reduction per row ----
        v8f acc0 = {}, acc1 = {};
        {
            const _Float16* krow = &Kh[l16 * DH];
            F16Frag kf0, kf1;
            kf0.q[0] = *(const uint4*)(krow + hi * 8);
            kf0.q[1] = *(const uint4*)(krow + hi * 8 + 16);
            kf1.q[0] = *(const uint4*)(krow + 32 + hi * 8);
            kf1.q[1] = *(const uint4*)(krow + 32 + hi * 8 + 16);
            acc0 = __builtin_amdgcn_wmma_f32_16x16x32_f16(false, qf[0].v, false, kf0.v,
                                                          (short)0, acc0, false, false);
            acc0 = __builtin_amdgcn_wmma_f32_16x16x32_f16(false, qf[1].v, false, kf1.v,
                                                          (short)0, acc0, false, false);
        }
        {
            const _Float16* krow = &Kh[(16 + l16) * DH];
            F16Frag kf0, kf1;
            kf0.q[0] = *(const uint4*)(krow + hi * 8);
            kf0.q[1] = *(const uint4*)(krow + hi * 8 + 16);
            kf1.q[0] = *(const uint4*)(krow + 32 + hi * 8);
            kf1.q[1] = *(const uint4*)(krow + 32 + hi * 8 + 16);
            acc1 = __builtin_amdgcn_wmma_f32_16x16x32_f16(false, qf[0].v, false, kf0.v,
                                                          (short)0, acc1, false, false);
            acc1 = __builtin_amdgcn_wmma_f32_16x16x32_f16(false, qf[1].v, false, kf1.v,
                                                          (short)0, acc1, false, false);
        }
        const int colg0 = kb + l16;
        const int colg1 = colg0 + 16;
        const int mv0 = am[colg0];
        const int mv1 = am[colg1];
        #pragma unroll
        for (int r = 0; r < 8; ++r) {
            const int rowg = m0 + r + hi * 8;       // C layout: lanes16-31 -> M+8
            float s0 = acc0[r]; if (colg0 > rowg || mv0 == 0) s0 = NEGV;
            float s1 = acc1[r]; if (colg1 > rowg || mv1 == 0) s1 = NEGV;
            float tmax = fmaxf(s0, s1);              // reduce 32 keys in one pass
            #pragma unroll
            for (int m = 1; m < 16; m <<= 1) tmax = fmaxf(tmax, __shfl_xor(tmax, m, 16));
            const float nmax = fmaxf(rmax[r], tmax);
            float e = __expf(s0 - nmax) + __expf(s1 - nmax);
            #pragma unroll
            for (int m = 1; m < 16; m <<= 1) e += __shfl_xor(e, m, 16);
            rsum[r] = rsum[r] * __expf(rmax[r] - nmax) + e;
            rmax[r] = nmax;
        }
    }

    float inv[8];
    #pragma unroll
    for (int r = 0; r < 8; ++r) inv[r] = (rsum[r] > 0.f) ? 1.f / rsum[r] : 0.f;

    // ================= pass B: write attn, accumulate O = P @ V ============
    v8f oacc[4] = {};                         // four 16-wide d chunks of O

    for (int p = 0; p < npairs; ++p) {
        const int kb = p * 32;
        __syncthreads();
        for (int idx = tid; idx < 32 * 16; idx += 128) {
            const int row = idx >> 4;
            const int c4  = (idx & 15) << 2;
            const float4 f = *(const float4*)(kp + (size_t)(kb + row) * DH + c4);
            _Float16* d = &Kh[row * DH + c4];
            d[0] = (_Float16)f.x; d[1] = (_Float16)f.y;
            d[2] = (_Float16)f.z; d[3] = (_Float16)f.w;
            const float4 g = *(const float4*)(vp + (size_t)(kb + row) * DH + c4);
            VTh[(c4 + 0) * 32 + row] = (_Float16)g.x;   // transpose V into [d][key]
            VTh[(c4 + 1) * 32 + row] = (_Float16)g.y;
            VTh[(c4 + 2) * 32 + row] = (_Float16)g.z;
            VTh[(c4 + 3) * 32 + row] = (_Float16)g.w;
        }
        if (p + 1 < npairs) {
            __builtin_prefetch(kp + (size_t)(kb + 32) * DH + tid * 16, 0, 3);
            __builtin_prefetch(vp + (size_t)(kb + 32) * DH + tid * 16, 0, 3);
        }
        __syncthreads();

        if (kb > mlast) {
            // fully masked for this wave's rows: stream zeros, skip compute
            #pragma unroll
            for (int s = 0; s < 2; ++s) {
                const int colg = kb + s * 16 + l16;
                #pragma unroll
                for (int r = 0; r < 8; ++r) {
                    const int rowg = m0 + r + hi * 8;
                    __builtin_nontemporal_store(0.f, attnp + (size_t)rowg * S_LEN + colg);
                }
            }
            continue;
        }

        #pragma unroll
        for (int s = 0; s < 2; ++s) {
            const _Float16* krow = &Kh[(s * 16 + l16) * DH];
            F16Frag kf0, kf1;
            kf0.q[0] = *(const uint4*)(krow + hi * 8);
            kf0.q[1] = *(const uint4*)(krow + hi * 8 + 16);
            kf1.q[0] = *(const uint4*)(krow + 32 + hi * 8);
            kf1.q[1] = *(const uint4*)(krow + 32 + hi * 8 + 16);
            v8f acc = {};
            acc = __builtin_amdgcn_wmma_f32_16x16x32_f16(false, qf[0].v, false, kf0.v,
                                                         (short)0, acc, false, false);
            acc = __builtin_amdgcn_wmma_f32_16x16x32_f16(false, qf[1].v, false, kf1.v,
                                                         (short)0, acc, false, false);
            const int colg = kb + s * 16 + l16;
            const int mv   = am[colg];
            #pragma unroll
            for (int r = 0; r < 8; ++r) {
                const int rowg = m0 + r + hi * 8;
                float sc = acc[r];
                if (colg > rowg || mv == 0) sc = NEGV;
                const float pe = __expf(sc - rmax[r]) * inv[r];   // masked -> exact 0
                // streamed, write-once: non-temporal keeps L2 for K/V
                __builtin_nontemporal_store(pe, attnp + (size_t)rowg * S_LEN + colg);
                Ph[wave][(r + hi * 8) * 32 + s * 16 + l16] = (_Float16)pe;
            }
        }

        // P A-frag from per-wave LDS scratch (same-wave DS ops are in order)
        F16Frag pf;
        const _Float16* prow = &Ph[wave][l16 * 32];
        pf.q[0] = *(const uint4*)(prow + hi * 8);
        pf.q[1] = *(const uint4*)(prow + hi * 8 + 16);
        #pragma unroll
        for (int nt = 0; nt < 4; ++nt) {
            const _Float16* vrow = &VTh[(nt * 16 + l16) * 32];
            F16Frag vf;
            vf.q[0] = *(const uint4*)(vrow + hi * 8);
            vf.q[1] = *(const uint4*)(vrow + hi * 8 + 16);
            oacc[nt] = __builtin_amdgcn_wmma_f32_16x16x32_f16(false, pf.v, false, vf.v,
                                                              (short)0, oacc[nt], false, false);
        }
    }

    // ---- store O (write-once -> non-temporal) ----
    #pragma unroll
    for (int nt = 0; nt < 4; ++nt) {
        #pragma unroll
        for (int r = 0; r < 8; ++r) {
            __builtin_nontemporal_store(oacc[nt][r],
                outp + (size_t)(m0 + r + hi * 8) * DH + nt * 16 + l16);
        }
    }

    // ---- zero-fill the untouched strictly-causal tail of attn ----
    const int colstart = npairs * 32;        // == r0 + 64, rows r0..r0+63 < colstart
    const int ncols = S_LEN - colstart;
    if (ncols > 0) {
        const int cpr = ncols >> 2;          // v4f per row
        const v4f z = {0.f, 0.f, 0.f, 0.f};
        for (int idx = tid; idx < 64 * cpr; idx += 128) {
            const int row = idx / cpr;
            const int c   = idx - row * cpr;
            __builtin_nontemporal_store(z,
                (v4f*)(attnp + (size_t)(r0 + row) * S_LEN + colstart + c * 4));
        }
    }
}

extern "C" void kernel_launch(void* const* d_in, const int* in_sizes, int n_in,
                              void* d_out, int out_size, void* d_ws, size_t ws_size,
                              hipStream_t stream) {
    (void)in_sizes; (void)n_in; (void)out_size; (void)d_ws; (void)ws_size;
    const float* q  = (const float*)d_in[0];
    const float* k  = (const float*)d_in[1];
    const float* v  = (const float*)d_in[2];
    const int*   am = (const int*)d_in[3];
    float* out  = (float*)d_out;                                   // [B,H,S,D]
    float* attn = out + (size_t)NBATCH * NHEAD * S_LEN * DH;       // [B,H,S,S]
    dim3 grid(S_LEN / 64, NBATCH * NHEAD);
    fa_fwd_attn_kernel<<<grid, 128, 0, stream>>>(q, k, v, am, out, attn);
}
